// TemporalIntentLSTM_44203803410894
// MI455X (gfx1250) — matrix operand order repaired and checked by reading, not compiled, not run on hardware
//
#include <hip/hip_runtime.h>

// Problem constants (match reference): B=256, C=4, T=2048, H=64
#define Bv 256
#define Tv 2048
#define Hv 64

typedef __attribute__((ext_vector_type(16))) _Float16 v16h;
typedef __attribute__((ext_vector_type(8)))  _Float16 v8h;
typedef __attribute__((ext_vector_type(8)))  float    v8f;

__device__ __forceinline__ float sigmoidf_fast(float x) {
    return 1.0f / (1.0f + __expf(-x));
}
__device__ __forceinline__ float tanhf_fast(float x) {
    return 2.0f / (1.0f + __expf(-2.0f * x)) - 1.0f;
}

// ---- gfx1250 async global->LDS copies (ASYNCcnt-tracked, ISA 08_async_tensor) ----
// LDS operand = 32-bit LDS byte address = low 32 bits of the generic pointer
// (LDS aperture keeps the offset in addr[31:0]).
__device__ __forceinline__ void async_copy_b128(void* dst_lds, const void* src_global) {
    uint32_t lds = (uint32_t)(uintptr_t)dst_lds;
    asm volatile("global_load_async_to_lds_b128 %0, %1, off"
                 : : "v"(lds), "v"(src_global) : "memory");
}
__device__ __forceinline__ void async_copy_b32(void* dst_lds, const void* src_global) {
    uint32_t lds = (uint32_t)(uintptr_t)dst_lds;
    asm volatile("global_load_async_to_lds_b32 %0, %1, off"
                 : : "v"(lds), "v"(src_global) : "memory");
}
__device__ __forceinline__ void wait_async_le1() {
    asm volatile("s_wait_asynccnt 0x1" ::: "memory");
}
__device__ __forceinline__ void wait_async_0() {
    asm volatile("s_wait_asynccnt 0x0" ::: "memory");
}

// A-fragment (16x32 f16, MxK) from LDS, per ISA 7.12.2:
// lane 0-15 = rows M=0..15, holds K = [0..7] then [16..23];
// lane 16-31 = same rows,   holds K = [8..15] then [24..31].
__device__ __forceinline__ v16h load_a_frag(const _Float16* base, int rowStride,
                                            int kt, int lane) {
    const int row  = lane & 15;
    const int half = lane >> 4;
    const _Float16* p = base + row * rowStride + kt * 32 + 8 * half;
    union { v16h v; v8h h[2]; } u;
    u.h[0] = *(const v8h*)(p);
    u.h[1] = *(const v8h*)(p + 16);
    return u.v;
}

// B-fragment (32x16 f16, KxN) gathered from global f32 weights W[n][k].
// Column N = lane%16, K = [8h..8h+7] then [16+8h..16+8h+7], h = lane/16.
__device__ __forceinline__ v16h load_b_frag_g(const float* __restrict__ w, int Kdim,
                                              int nt, int kt, int lane) {
    const int n  = nt * 16 + (lane & 15);
    const int kb = kt * 32 + (lane >> 4) * 8;
    const float* p = w + n * Kdim + kb;
    v16h r;
#pragma unroll
    for (int j = 0; j < 16; ++j) r[j] = (_Float16)p[j < 8 ? j : j + 8];
    return r;
}

// One bidirectional-LSTM layer. blockIdx.x = batch tile of 16, blockIdx.y = direction.
// D==4  : layer 0, input x[B,C,T] f32, input GEMM via 4 scalar FMAs (weights in VGPRs).
// D==128: layer 1, input hin[B,T,128] f16, input GEMM via WMMA, tiles staged with
//         double-buffered GLOBAL_LOAD_ASYNC_TO_LDS.
// Output hout[B,T,128] f16 at column offset 0 (fwd) or 64 (rev).
template <int D>
__global__ __launch_bounds__(256)
void lstm_layer_kernel(const float* __restrict__ x,
                       const _Float16* __restrict__ hin,
                       const float* __restrict__ wih_f, const float* __restrict__ whh_f,
                       const float* __restrict__ bih_f, const float* __restrict__ bhh_f,
                       const float* __restrict__ wih_r, const float* __restrict__ whh_r,
                       const float* __restrict__ bih_r, const float* __restrict__ bhh_r,
                       _Float16* __restrict__ hout) {
    constexpr int KTIN = D / 32;  // 0 for D==4 (scalar path), 4 for D==128
    const int tid  = threadIdx.x;
    const int lane = tid & 31;
    const int wv   = tid >> 5;       // wave 0..7, owns N-tiles 2w, 2w+1
    const int b0   = blockIdx.x * 16;
    const bool rev = (blockIdx.y != 0);

    const float* wih = rev ? wih_r : wih_f;
    const float* whh = rev ? whh_r : whh_f;
    const float* bih = rev ? bih_r : bih_f;
    const float* bhh = rev ? bhh_r : bhh_f;
    const int col_off = rev ? Hv : 0;

    __shared__ float    g_buf[16 * 256];     // gate values [m][n]
    __shared__ _Float16 sH[16 * Hv];         // current hidden state (A operand)
    __shared__ _Float16 sXb[2][16 * 128];    // double-buffered layer-1 input tile
    __shared__ float    sBias[256];
    __shared__ float    sXin[2][64];         // double-buffered layer-0 input tile

    // ---- Stationary weights live in VGPRs for the whole scan ----
    v16h bWhh[2][2];                        // [kt][p] recurrent, K=64
    v16h bWih[(KTIN > 0 ? KTIN : 1)][2];    // [kt][p] input, K=128 (layer 1)
    float wihReg[2][4];                     // layer-0 input weights, 2 gate rows/lane
#pragma unroll
    for (int p = 0; p < 2; ++p) {
        const int nt = wv * 2 + p;
#pragma unroll
        for (int kt = 0; kt < 2; ++kt)
            bWhh[kt][p] = load_b_frag_g(whh, Hv, nt, kt, lane);
#pragma unroll
        for (int kt = 0; kt < KTIN; ++kt)
            bWih[kt][p] = load_b_frag_g(wih, D, nt, kt, lane);
        if (D == 4) {
            const int ng = nt * 16 + (lane & 15);
#pragma unroll
            for (int d = 0; d < 4; ++d) wihReg[p][d] = wih[ng * 4 + d];
        }
    }

    for (int i = tid; i < 256; i += 256) sBias[i] = bih[i] + bhh[i];
    for (int i = tid; i < 16 * Hv; i += 256) sH[i] = (_Float16)0.0f;

    // Cell state: 4 cells per thread, m = tid%16, j = 4*(tid/16)+q
    float cst[4] = {0.f, 0.f, 0.f, 0.f};
    const int um = tid & 15;
    const int uj = (tid >> 4) * 4;

    // Per-thread staging assignment
    const int srow = tid >> 4, sseg = tid & 15;   // layer 1: 16 rows x 16B chunks
    const int xb   = b0 + (tid >> 2), xd = tid & 3;  // layer 0: 64 scalars

    // ---- Preload timestep 0 tile asynchronously ----
    {
        const int t0 = rev ? (Tv - 1) : 0;
        if (D == 128) {
            async_copy_b128(&sXb[0][srow * 128 + sseg * 8],
                            hin + ((size_t)(b0 + srow) * Tv + t0) * 128 + sseg * 8);
        } else if (tid < 64) {
            async_copy_b32(&sXin[0][tid], x + ((size_t)xb * 4 + xd) * Tv + t0);
        }
    }
    __syncthreads();  // sH zero-init visible

    for (int s = 0; s < Tv; ++s) {
        const int t   = rev ? (Tv - 1 - s) : s;
        const int cur = s & 1;

        // ---- Phase A: kick off async copy of the NEXT tile, wait for current ----
        if (s + 1 < Tv) {
            const int tn = rev ? (Tv - 2 - s) : (s + 1);
            if (D == 128) {
                async_copy_b128(&sXb[cur ^ 1][srow * 128 + sseg * 8],
                                hin + ((size_t)(b0 + srow) * Tv + tn) * 128 + sseg * 8);
            } else if (tid < 64) {
                async_copy_b32(&sXin[cur ^ 1][tid], x + ((size_t)xb * 4 + xd) * Tv + tn);
            }
            wait_async_le1();   // in-order: current tile resident for this wave
        } else {
            wait_async_0();
        }
        __syncthreads();        // all waves' async data + prev sH visible

        // ---- Phase B: gate GEMM via WMMA; g = x@Wih^T + h@Whh^T + bias ----
        // A fragments are shared by both N-tiles: load once.
        v16h aH[2];
#pragma unroll
        for (int kt = 0; kt < 2; ++kt) aH[kt] = load_a_frag(sH, Hv, kt, lane);
        v16h aX[(KTIN > 0 ? KTIN : 1)];
#pragma unroll
        for (int kt = 0; kt < KTIN; ++kt) aX[kt] = load_a_frag(sXb[cur], 128, kt, lane);

#pragma unroll
        for (int p = 0; p < 2; ++p) {
            v8f acc = {};
#pragma unroll
            for (int kt = 0; kt < KTIN; ++kt)   // layer-1 input GEMM, K=128
                acc = __builtin_amdgcn_wmma_f32_16x16x32_f16(
                    false, aX[kt], false, bWih[kt][p], (short)0, acc, false, false);
#pragma unroll
            for (int kt = 0; kt < 2; ++kt)      // recurrent GEMM, K=64
                acc = __builtin_amdgcn_wmma_f32_16x16x32_f16(
                    false, aH[kt], false, bWhh[kt][p], (short)0, acc, false, false);

            // D-matrix layout: M = r + 8*(lane/16), N = nt*16 + lane%16
            const int nl = lane & 15;
            const int mb = (lane >> 4) * 8;
            const int ng = (wv * 2 + p) * 16 + nl;
#pragma unroll
            for (int r = 0; r < 8; ++r) {
                float v = acc[r] + sBias[ng];
                if (D == 4) {  // layer-0 input GEMM, K=4: weights held in VGPRs
#pragma unroll
                    for (int d = 0; d < 4; ++d)
                        v += sXin[cur][(mb + r) * 4 + d] * wihReg[p][d];
                }
                g_buf[(mb + r) * 256 + ng] = v;
            }
        }
        __syncthreads();  // g_buf complete; sH/sX reads done

        // ---- Phase C: nonlinearities + state update (PyTorch i,f,g,o order) ----
#pragma unroll
        for (int q = 0; q < 4; ++q) {
            const int j  = uj + q;
            const float gi = sigmoidf_fast(g_buf[um * 256 + j]);
            const float gf = sigmoidf_fast(g_buf[um * 256 + 64 + j]);
            const float gg = tanhf_fast(g_buf[um * 256 + 128 + j]);
            const float go = sigmoidf_fast(g_buf[um * 256 + 192 + j]);
            const float c  = gf * cst[q] + gi * gg;
            cst[q] = c;
            const float h = go * tanhf_fast(c);
            sH[um * Hv + j] = (_Float16)h;
            hout[((size_t)(b0 + um) * Tv + t) * 128 + col_off + j] = (_Float16)h;
        }
        // next Phase A writes sXb[cur^1]/sXin[cur^1]; its last readers finished
        // before the post-Phase-B barrier of this step -> safe without extra barrier
    }
}

// out[b,t,:] = h1[b,t,:] @ fc_w^T + fc_b   (memory-bound; one thread per (b,t))
__global__ __launch_bounds__(256)
void fc_kernel(const _Float16* __restrict__ h1, const float* __restrict__ fcw,
               const float* __restrict__ fcb, float* __restrict__ out) {
    const int idx = blockIdx.x * blockDim.x + threadIdx.x;  // b*T + t
    const _Float16* row = h1 + (size_t)idx * 128;
    float a0 = fcb[0], a1 = fcb[1];
#pragma unroll 4
    for (int j = 0; j < 128; j += 8) {
        v8h v = *(const v8h*)(row + j);
#pragma unroll
        for (int k = 0; k < 8; ++k) {
            const float f = (float)v[k];
            a0 += f * fcw[j + k];
            a1 += f * fcw[128 + j + k];
        }
    }
    out[(size_t)idx * 2 + 0] = a0;
    out[(size_t)idx * 2 + 1] = a1;
}

extern "C" void kernel_launch(void* const* d_in, const int* in_sizes, int n_in,
                              void* d_out, int out_size, void* d_ws, size_t ws_size,
                              hipStream_t stream) {
    (void)in_sizes; (void)n_in; (void)out_size; (void)ws_size;
    const float* x         = (const float*)d_in[0];
    const float* w_ih_l0   = (const float*)d_in[1];
    const float* w_hh_l0   = (const float*)d_in[2];
    const float* b_ih_l0   = (const float*)d_in[3];
    const float* b_hh_l0   = (const float*)d_in[4];
    const float* w_ih_l0r  = (const float*)d_in[5];
    const float* w_hh_l0r  = (const float*)d_in[6];
    const float* b_ih_l0r  = (const float*)d_in[7];
    const float* b_hh_l0r  = (const float*)d_in[8];
    const float* w_ih_l1   = (const float*)d_in[9];
    const float* w_hh_l1   = (const float*)d_in[10];
    const float* b_ih_l1   = (const float*)d_in[11];
    const float* b_hh_l1   = (const float*)d_in[12];
    const float* w_ih_l1r  = (const float*)d_in[13];
    const float* w_hh_l1r  = (const float*)d_in[14];
    const float* b_ih_l1r  = (const float*)d_in[15];
    const float* b_hh_l1r  = (const float*)d_in[16];
    const float* fc_w      = (const float*)d_in[17];
    const float* fc_b      = (const float*)d_in[18];

    // Workspace: h0 [B,T,128] f16 (128 MB) then h1 [B,T,128] f16 (128 MB).
    _Float16* h0 = (_Float16*)d_ws;
    _Float16* h1 = h0 + (size_t)Bv * Tv * 128;

    const dim3 blk(256);
    const dim3 grd(Bv / 16, 2);  // 16 batch tiles x 2 directions

    lstm_layer_kernel<4><<<grd, blk, 0, stream>>>(
        x, nullptr,
        w_ih_l0, w_hh_l0, b_ih_l0, b_hh_l0,
        w_ih_l0r, w_hh_l0r, b_ih_l0r, b_hh_l0r, h0);

    lstm_layer_kernel<128><<<grd, blk, 0, stream>>>(
        nullptr, h0,
        w_ih_l1, w_hh_l1, b_ih_l1, b_hh_l1,
        w_ih_l1r, w_hh_l1r, b_ih_l1r, b_hh_l1r, h1);

    fc_kernel<<<(Bv * Tv) / 256, blk, 0, stream>>>(h1, fc_w, fc_b, (float*)d_out);
}
